// GQA_Triton_59785944760336
// MI455X (gfx1250) — matrix operand order repaired and checked by reading, not compiled, not run on hardware
//
#include <hip/hip_runtime.h>
#include <hip/hip_bf16.h>

typedef __attribute__((ext_vector_type(16))) _Float16 v16h;
typedef __attribute__((ext_vector_type(8)))  _Float16 v8h;
typedef __attribute__((ext_vector_type(8)))  float    v8f;
typedef __attribute__((ext_vector_type(4)))  float    v4f;

#define S_LEN   2048
#define D_IN    2048
#define NH      32
#define NKVH    8
#define HD      64
#define GROUPS  4
#define BS      2
#define SCALE   0.125f
#define EPS     1e-6f

__device__ inline v8f wmma_f16(v16h a, v16h b, v8f c) {
  return __builtin_amdgcn_wmma_f32_16x16x32_f16(false, a, false, b, (short)0, c, false, false);
}

__device__ inline v16h frag16(const _Float16* p0, const _Float16* p1) {
  v8h lo = *(const v8h*)p0;
  v8h hi = *(const v8h*)p1;
  return __builtin_shufflevector(lo, hi, 0,1,2,3,4,5,6,7,8,9,10,11,12,13,14,15);
}

__device__ inline v8f vzero8() {
  v8f z;
#pragma unroll
  for (int i = 0; i < 8; ++i) z[i] = 0.0f;
  return z;
}

// 16B-per-op async copy global(f16) -> LDS, tracked by ASYNCcnt.
__device__ inline void async_cp16B(const _Float16* g, const _Float16* l) {
  const unsigned loff = (unsigned)(size_t)l;   // low 32 bits = LDS offset
  asm volatile("global_load_async_to_lds_b128 %0, %1, off"
               :: "v"(loff), "v"(g) : "memory");
}
__device__ inline void async_wait0() {
  asm volatile("s_wait_asynccnt 0x0" ::: "memory");
}

// load 16 consecutive fp32 from global, convert, store as 2 x v8h to LDS
__device__ inline void cvt16(const float* __restrict__ g, _Float16* s) {
  const v4f* gv = (const v4f*)g;
  v4f f0 = gv[0], f1 = gv[1], f2 = gv[2], f3 = gv[3];
  v8h h0, h1;
  h0[0]=(_Float16)f0[0]; h0[1]=(_Float16)f0[1]; h0[2]=(_Float16)f0[2]; h0[3]=(_Float16)f0[3];
  h0[4]=(_Float16)f1[0]; h0[5]=(_Float16)f1[1]; h0[6]=(_Float16)f1[2]; h0[7]=(_Float16)f1[3];
  h1[0]=(_Float16)f2[0]; h1[1]=(_Float16)f2[1]; h1[2]=(_Float16)f2[2]; h1[3]=(_Float16)f2[3];
  h1[4]=(_Float16)f3[0]; h1[5]=(_Float16)f3[1]; h1[6]=(_Float16)f3[2]; h1[7]=(_Float16)f3[3];
  *(v8h*)s       = h0;
  *(v8h*)(s + 8) = h1;
}

// ---------------------------------------------------------------------------
// C[M,N] = A[M,K] * B[N,K]^T  (NT GEMM), fp32 in/out, f16 WMMA compute.
// block tile 256x128, K-step 32, 256 threads = 8 waves (4x2), wave tile 64x64.
// M multiple of 256; N multiple of 128; K multiple of 32.
// ---------------------------------------------------------------------------
__global__ __launch_bounds__(256) void gemm_nt_wmma(
    const float* __restrict__ A, int lda,
    const float* __restrict__ B, int ldb,
    float* __restrict__ C, int ldc, int K) {
  __shared__ __attribute__((aligned(16))) _Float16 sA[256 * 40];
  __shared__ __attribute__((aligned(16))) _Float16 sB[128 * 40];

  const int tid  = threadIdx.x;
  const int lane = tid & 31;
  const int w    = tid >> 5;
  const int wm   = w & 3;        // 0..3 -> 64-row slab
  const int wn   = w >> 2;       // 0..1 -> 64-col slab
  const int m0   = blockIdx.y * 256;
  const int n0   = blockIdx.x * 128;

  const int half = lane >> 4;
  const int l16  = lane & 15;
  const int kb0  = half * 8;     // A-fragment K base
  const int kbB  = half * 8;     // B-fragment K base

  const int lr = tid >> 1;          // 0..127 tile row for coop loads
  const int lc = (tid & 1) * 16;    // 0 or 16

  v8f acc[4][4];
#pragma unroll
  for (int i = 0; i < 4; ++i)
#pragma unroll
    for (int j = 0; j < 4; ++j) acc[i][j] = vzero8();

  for (int kt = 0; kt < K; kt += 32) {
    cvt16(A + (size_t)(m0 + lr) * lda + kt + lc,       &sA[lr * 40 + lc]);
    cvt16(A + (size_t)(m0 + lr + 128) * lda + kt + lc, &sA[(lr + 128) * 40 + lc]);
    cvt16(B + (size_t)(n0 + lr) * ldb + kt + lc,       &sB[lr * 40 + lc]);
    if (kt + 32 < K) {
      __builtin_prefetch(A + (size_t)(m0 + lr) * lda + kt + 32 + lc, 0, 1);
      __builtin_prefetch(A + (size_t)(m0 + lr + 128) * lda + kt + 32 + lc, 0, 1);
      __builtin_prefetch(B + (size_t)(n0 + lr) * ldb + kt + 32 + lc, 0, 1);
    }
    __syncthreads();

    v16h aF[4], bF[4];
#pragma unroll
    for (int ms = 0; ms < 4; ++ms) {
      const int r = wm * 64 + ms * 16 + l16;
      aF[ms] = frag16(&sA[r * 40 + kb0], &sA[r * 40 + kb0 + 16]);
    }
#pragma unroll
    for (int ns = 0; ns < 4; ++ns) {
      const int r = wn * 64 + ns * 16 + l16;
      bF[ns] = frag16(&sB[r * 40 + kbB], &sB[r * 40 + kbB + 16]);
    }
#pragma unroll
    for (int ms = 0; ms < 4; ++ms)
#pragma unroll
      for (int ns = 0; ns < 4; ++ns)
        acc[ms][ns] = wmma_f16(aF[ms], bF[ns], acc[ms][ns]);

    __syncthreads();
  }

#pragma unroll
  for (int ms = 0; ms < 4; ++ms)
#pragma unroll
    for (int ns = 0; ns < 4; ++ns) {
      const int gm = m0 + wm * 64 + ms * 16 + half * 8;
      const int gn = n0 + wn * 64 + ns * 16 + l16;
      float* pc = C + (size_t)gm * ldc + gn;
#pragma unroll
      for (int i = 0; i < 8; ++i) pc[(size_t)i * ldc] = acc[ms][ns][i];
    }
}

// ---------------------------------------------------------------------------
// RMSNorm + RoPE + pack to f16 [B, heads, S, HD].
// One wave handles one (token, head-slot); slots: 0..31=Q, 32..39=K, 40..47=V.
// ---------------------------------------------------------------------------
__global__ __launch_bounds__(256) void normrope(
    const float* __restrict__ qkv,   // [B*S, 3072]
    const float* __restrict__ cosb,  // [S, 64]
    const float* __restrict__ sinb,  // [S, 64]
    const float* __restrict__ wqn,   // [64]
    const float* __restrict__ wkn,   // [64]
    _Float16* __restrict__ Qh,       // [B, NH,  S, 64]
    _Float16* __restrict__ Kh,       // [B, NKVH,S, 64]
    _Float16* __restrict__ Vh) {     // [B, NKVH,S, 64]
  const int g     = blockIdx.x * 8 + (threadIdx.x >> 5);
  const int lane  = threadIdx.x & 31;
  const int token = g / 48;
  const int hh    = g % 48;
  const int b     = token >> 11;       // /S_LEN
  const int s     = token & (S_LEN - 1);
  const float* row = qkv + (size_t)token * 3072;
  const int d0 = lane, d1 = lane + 32;

  if (hh < 40) {  // Q or K head: RMSNorm + RoPE
    const bool isQ = (hh < 32);
    const int  hd  = isQ ? hh : (hh - 32);
    const float* p = row + (isQ ? hd * 64 : 2048 + hd * 64);
    const float* wn = isQ ? wqn : wkn;
    float t0 = p[d0], t1 = p[d1];
    float ss = t0 * t0 + t1 * t1;
#pragma unroll
    for (int d = 16; d >= 1; d >>= 1) ss += __shfl_xor(ss, d, 32);
    const float inv = rsqrtf(ss * (1.0f / 64.0f) + EPS);
    const float n0 = t0 * inv * wn[d0];
    const float n1 = t1 * inv * wn[d1];
    const float c0 = cosb[s * 64 + d0], c1 = cosb[s * 64 + d1];
    const float s0 = sinb[s * 64 + d0], s1 = sinb[s * 64 + d1];
    const float r0 = n0 * c0 - n1 * s0;    // d < 32: -second_half * sin
    const float r1 = n1 * c1 + n0 * s1;    // d >=32: +first_half * sin
    _Float16* o = (isQ ? Qh + ((size_t)(b * NH + hd) * S_LEN + s) * HD
                       : Kh + ((size_t)(b * NKVH + hd) * S_LEN + s) * HD);
    o[d0] = (_Float16)r0;
    o[d1] = (_Float16)r1;
  } else {        // V head: convert + transpose only
    const int kvh = hh - 40;
    const float* p = row + 2560 + kvh * 64;
    _Float16* o = Vh + ((size_t)(b * NKVH + kvh) * S_LEN + s) * HD;
    o[d0] = (_Float16)p[d0];
    o[d1] = (_Float16)p[d1];
  }
}

// ---------------------------------------------------------------------------
// Causal flash attention. grid=(B*NH, S/64), block=128 (4 waves).
// Each wave owns 16 query rows; K tile arrives via GLOBAL_LOAD_ASYNC_TO_LDS,
// V tile is transposed into LDS so P*V B-fragments are contiguous.
// Output: attn [B, S, NH*HD] fp32.
// ---------------------------------------------------------------------------
__global__ __launch_bounds__(128) void fa_kernel(
    const _Float16* __restrict__ Qh,
    const _Float16* __restrict__ Kh,
    const _Float16* __restrict__ Vh,
    float* __restrict__ O) {
  __shared__ __attribute__((aligned(16))) _Float16 sK[32 * 72];       // [key][d]
  __shared__ __attribute__((aligned(16))) _Float16 sVt[64 * 40];      // [d][key]
  __shared__ __attribute__((aligned(16))) _Float16 sP[4 * 16 * 40];   // per-wave P

  const int tid  = threadIdx.x;
  const int lane = tid & 31;
  const int w    = tid >> 5;
  const int bh   = blockIdx.x;
  const int b    = bh >> 5;            // /NH
  const int h    = bh & 31;
  const int kvh  = h >> 2;             // /GROUPS
  const int q0   = blockIdx.y * 64;
  const int qw   = q0 + w * 16;

  const int half = lane >> 4;
  const int l16  = lane & 15;
  const int kb0  = half * 8;
  const int kbB  = half * 8;

  // Q fragments (16 rows x 64 d = two K-chunks), loaded straight from global
  const _Float16* qbase = Qh + ((size_t)(b * NH + h) * S_LEN + qw) * HD;
  v16h aQ[2];
#pragma unroll
  for (int kt = 0; kt < 2; ++kt) {
    const _Float16* p = qbase + (size_t)l16 * HD + kt * 32 + kb0;
    aQ[kt] = frag16(p, p + 16);
  }

  v8f o[4];
#pragma unroll
  for (int d = 0; d < 4; ++d) o[d] = vzero8();
  float m_r[8], l_r[8];
#pragma unroll
  for (int i = 0; i < 8; ++i) { m_r[i] = -3.0e38f; l_r[i] = 0.0f; }

  const _Float16* kg = Kh + (size_t)(b * NKVH + kvh) * S_LEN * HD;
  const _Float16* vg = Vh + (size_t)(b * NKVH + kvh) * S_LEN * HD;

  const int krow = tid >> 2;          // 0..31
  const int kcol = (tid & 3) * 16;    // 0,16,32,48
  const int ntiles = q0 / 32 + 2;     // keys through q0+63 (causal)

  for (int t = 0; t < ntiles; ++t) {
    const int k0 = t * 32;
    // K tile: 4KB memory->LDS via async copy (2 x 16B chunks per thread)
#pragma unroll
    for (int j = 0; j < 2; ++j) {
      const int c   = tid + j * 128;          // 0..255 chunk id
      const int row = c >> 3;                 // 0..31
      const int col = (c & 7) * 8;            // 0..56
      async_cp16B(kg + (size_t)(k0 + row) * HD + col, &sK[row * 72 + col]);
    }
    // V tile: manual transpose scatter into sVt (overlaps with async K)
    {
      const _Float16* gv = vg + (size_t)(k0 + krow) * HD + kcol;
      v8h y0 = *(const v8h*)gv, y1 = *(const v8h*)(gv + 8);
#pragma unroll
      for (int j = 0; j < 8; ++j) {
        sVt[(kcol + j) * 40 + krow]     = y0[j];
        sVt[(kcol + 8 + j) * 40 + krow] = y1[j];
      }
    }
    async_wait0();
    __syncthreads();

    // scores: two 16-key subtiles, each = 2 WMMAs over d
    v8f s[2];
#pragma unroll
    for (int ns = 0; ns < 2; ++ns) {
      const int kr = ns * 16 + l16;
      v16h b0 = frag16(&sK[kr * 72 + kbB],      &sK[kr * 72 + kbB + 16]);
      v16h b1 = frag16(&sK[kr * 72 + 32 + kbB], &sK[kr * 72 + 32 + kbB + 16]);
      v8f c = vzero8();
      c = wmma_f16(aQ[0], b0, c);
      c = wmma_f16(aQ[1], b1, c);
      const int kidx = k0 + ns * 16 + l16;
#pragma unroll
      for (int i = 0; i < 8; ++i) {
        const int qidx = qw + i + half * 8;
        s[ns][i] = (kidx <= qidx) ? c[i] * SCALE : -3.0e38f;
      }
    }

    // online softmax (row i+8*half spans 16 lanes of this half-wave)
    float alpha[8];
#pragma unroll
    for (int i = 0; i < 8; ++i) {
      float mx = fmaxf(s[0][i], s[1][i]);
#pragma unroll
      for (int d = 8; d >= 1; d >>= 1) mx = fmaxf(mx, __shfl_xor(mx, d, 32));
      const float mn = fmaxf(m_r[i], mx);
      alpha[i] = __expf(m_r[i] - mn);
      m_r[i] = mn;
      const float p0 = __expf(s[0][i] - mn);
      const float p1 = __expf(s[1][i] - mn);
      s[0][i] = p0; s[1][i] = p1;
      float rs = p0 + p1;
#pragma unroll
      for (int d = 8; d >= 1; d >>= 1) rs += __shfl_xor(rs, d, 32);
      l_r[i] = l_r[i] * alpha[i] + rs;
    }
#pragma unroll
    for (int d = 0; d < 4; ++d)
#pragma unroll
      for (int i = 0; i < 8; ++i) o[d][i] *= alpha[i];

    // stash P in LDS (f16) and reload as an A-fragment
    _Float16* pw = &sP[w * 16 * 40];
#pragma unroll
    for (int ns = 0; ns < 2; ++ns)
#pragma unroll
      for (int i = 0; i < 8; ++i)
        pw[(i + half * 8) * 40 + ns * 16 + l16] = (_Float16)s[ns][i];

    {
      const _Float16* pr = pw + l16 * 40 + kb0;
      v16h aP = frag16(pr, pr + 16);
#pragma unroll
      for (int ds = 0; ds < 4; ++ds) {
        const int dr = ds * 16 + l16;
        v16h bV = frag16(&sVt[dr * 40 + kbB], &sVt[dr * 40 + kbB + 16]);
        o[ds] = wmma_f16(aP, bV, o[ds]);
      }
    }
    __syncthreads();
  }

  // epilogue: normalize and write [B, S, NH*HD]
#pragma unroll
  for (int i = 0; i < 8; ++i) l_r[i] = 1.0f / l_r[i];
#pragma unroll
  for (int ds = 0; ds < 4; ++ds)
#pragma unroll
    for (int i = 0; i < 8; ++i) {
      const int q = qw + i + half * 8;
      O[((size_t)(b * S_LEN + q)) * (NH * HD) + h * HD + ds * 16 + l16] =
          o[ds][i] * l_r[i];
    }
}

// ---------------------------------------------------------------------------
extern "C" void kernel_launch(void* const* d_in, const int* in_sizes, int n_in,
                              void* d_out, int out_size, void* d_ws, size_t ws_size,
                              hipStream_t stream) {
  const float* x    = (const float*)d_in[0];
  // d_in[1] = mask (unused; causal hardcoded)
  const float* cosb = (const float*)d_in[2];
  const float* sinb = (const float*)d_in[3];
  const float* Wq   = (const float*)d_in[4];
  const float* Wk   = (const float*)d_in[5];
  const float* Wv   = (const float*)d_in[6];
  const float* Wo   = (const float*)d_in[7];
  const float* wqn  = (const float*)d_in[8];
  const float* wkn  = (const float*)d_in[9];
  float* out = (float*)d_out;

  char* ws = (char*)d_ws;
  const size_t M = (size_t)BS * S_LEN;                 // 4096 tokens
  float* qkv  = (float*)ws;                            // [4096, 3072] fp32
  float* attn = qkv;                                   // alias (qkv consumed first)
  size_t off = M * 3072 * sizeof(float);
  _Float16* Qh = (_Float16*)(ws + off); off += (size_t)BS * NH   * S_LEN * HD * 2;
  _Float16* Kh = (_Float16*)(ws + off); off += (size_t)BS * NKVH * S_LEN * HD * 2;
  _Float16* Vh = (_Float16*)(ws + off);

  dim3 blk(256);
  // QKV projections: qkv[:, 0:2048 | 2048:2560 | 2560:3072]
  gemm_nt_wmma<<<dim3(16, 16), blk, 0, stream>>>(x, D_IN, Wq, D_IN, qkv,        3072, D_IN);
  gemm_nt_wmma<<<dim3(4,  16), blk, 0, stream>>>(x, D_IN, Wk, D_IN, qkv + 2048, 3072, D_IN);
  gemm_nt_wmma<<<dim3(4,  16), blk, 0, stream>>>(x, D_IN, Wv, D_IN, qkv + 2560, 3072, D_IN);
  // RMSNorm + RoPE + pack to f16 head-major
  normrope<<<dim3((unsigned)(M * 48 / 8)), dim3(256), 0, stream>>>(
      qkv, cosb, sinb, wqn, wkn, Qh, Kh, Vh);
  // causal flash attention
  fa_kernel<<<dim3(BS * NH, S_LEN / 64), dim3(128), 0, stream>>>(Qh, Kh, Vh, attn);
  // output projection
  gemm_nt_wmma<<<dim3(16, 16), blk, 0, stream>>>(attn, NH * HD, Wo, NH * HD, out, D_IN, D_IN);
}
